// GraphConvLayer_11020886082334
// MI455X (gfx1250) — matrix operand order repaired and checked by reading, not compiled, not run on hardware
//
#include <hip/hip_runtime.h>
#include <hip/hip_bf16.h>
#include <math.h>

// ---------------------------------------------------------------------------
// GraphConv layer for MI455X (gfx1250, wave32, WMMA).
//   messages = prepareFFN(gather(node_reps)) * ew    (BN folded into weights)
//   agg      = segment_mean(messages)                (hw f32 atomics)
//   out      = updateFFN(concat(node_reps, agg))
// All GEMMs via v_wmma_f32_16x16x32_bf16 (f32 accumulate).
// GELU uses a branchless erf (A&S 7.1.26, |err|<1.5e-7) on v_rcp/v_exp
// transcendentals so it co-executes with the WMMA pipe instead of the branchy
// libm erff (which wraps EXEC-divergent regions around the matrix code).
// ---------------------------------------------------------------------------

typedef __attribute__((ext_vector_type(16))) __bf16 v16bf;
typedef __attribute__((ext_vector_type(8)))  float  v8f;

#define DIM 128
#define BN_EPS 1e-3f

union V16Q { uint4 q[2]; v16bf v; };

__device__ __forceinline__ float gelu_fast_exact(float x) {
  // erf(|x|/sqrt2) via Abramowitz-Stegun 7.1.26 (branchless, |err|<1.5e-7)
  const float ax = fabsf(x) * 0.70710678118654752440f;
  const float t  = __builtin_amdgcn_rcpf(fmaf(0.3275911f, ax, 1.0f));
  float p = 1.061405429f;
  p = fmaf(p, t, -1.453152027f);
  p = fmaf(p, t, 1.421413741f);
  p = fmaf(p, t, -0.284496736f);
  p = fmaf(p, t, 0.254829592f);
  const float e  = __builtin_amdgcn_exp2f(ax * ax * -1.4426950408889634f);
  const float er = fmaf(-p * t, e, 1.0f);          // erf(|x|/sqrt2)
  return 0.5f * x * (1.0f + copysignf(er, x));
}

// ---------------------------------------------------------------------------
// Fold BN into Dense and pack W' into wave32 B-fragment order.
//   fragment (kk, nn), lane l, elem i  ->  (((kk*8+nn)*32 + l)*16 + i)
//   holds W'[kk*32 + (i<8?0:16) + (l<16?0:8) + i%8][nn*16 + l%16]
// ---------------------------------------------------------------------------
__global__ __launch_bounds__(128) void pack_kernel(
    const float* __restrict__ gamma, const float* __restrict__ beta,
    const float* __restrict__ mu,    const float* __restrict__ var,
    const float* __restrict__ w,     const float* __restrict__ b,
    int din, __bf16* __restrict__ Wp, float* __restrict__ Bp) {
  __shared__ float s_s[256], s_t[256];
  const int j = threadIdx.x;  // output column 0..127
  for (int k = j; k < din; k += 128) {
    float s = gamma[k] * rsqrtf(var[k] + BN_EPS);
    s_s[k] = s;
    s_t[k] = beta[k] - mu[k] * s;
  }
  __syncthreads();
  float bacc = b[j];
  const int nn = j >> 4;
  for (int k = 0; k < din; ++k) {
    const float wv = w[k * DIM + j];
    bacc += s_t[k] * wv;
    const int kk   = k >> 5;
    const int kb   = k & 31;
    const int lane = (((kb >> 3) & 1) << 4) + (j & 15);
    const int i    = (kb & 7) + ((kb >> 4) << 3);
    Wp[(((kk * 8 + nn) * 32 + lane) << 4) + i] = (__bf16)(s_s[k] * wv);
  }
  Bp[j] = bacc;
}

__global__ void zero_kernel(float* __restrict__ sums, float* __restrict__ counts,
                            size_t nS, int nC) {
  const size_t stride = (size_t)gridDim.x * blockDim.x;
  for (size_t i = (size_t)blockIdx.x * blockDim.x + threadIdx.x; i < nS; i += stride)
    sums[i] = 0.0f;
  for (size_t i = (size_t)blockIdx.x * blockDim.x + threadIdx.x; i < (size_t)nC; i += stride)
    counts[i] = 0.0f;
}

// ---------------------------------------------------------------------------
// Message kernel: one wave per 16-edge tile. 64 WMMAs per tile (2 layers).
// ---------------------------------------------------------------------------
__global__ __launch_bounds__(128) void msg_kernel(
    const float* __restrict__ node_reps, const int* __restrict__ edges,
    const float* __restrict__ edge_weights,
    const __bf16* __restrict__ W1, const float* __restrict__ B1,
    const __bf16* __restrict__ W2, const float* __restrict__ B2,
    float* __restrict__ sums, float* __restrict__ counts, int nEdges) {
  __shared__ __align__(16) __bf16 act[4][16][DIM];  // per-wave staging
  const int wave = threadIdx.x >> 5;
  const int lane = threadIdx.x & 31;
  const int eb   = (blockIdx.x * 4 + wave) * 16;
  if (eb >= nEdges) return;
  const int half = (lane < 16) ? 0 : 8;             // A K-chunk / C M-row offset
  const int row  = lane & 15;

  __builtin_prefetch(W1 + (lane << 6), 0, 1);       // warm packed weights
  __builtin_prefetch(W2 + (lane << 6), 0, 1);

  // gather neighbour row into A fragments (16-bit A 16x32 layout)
  const int er  = (eb + row < nEdges) ? (eb + row) : (nEdges - 1);
  const int nbr = edges[nEdges + er];               // edges[1][e]
  const float* __restrict__ src = node_reps + (size_t)nbr * DIM;
  v16bf a[4];
#pragma unroll
  for (int kk = 0; kk < 4; ++kk) {
    const int kb = kk * 32 + half;
#pragma unroll
    for (int i = 0; i < 8; ++i) {
      a[kk][i]     = (__bf16)src[kb + i];
      a[kk][8 + i] = (__bf16)src[kb + 16 + i];
    }
  }

  // layer 1: [16x128] @ W1'[128x128] -> gelu -> LDS (bf16)
#pragma unroll
  for (int nn = 0; nn < 8; ++nn) {
    v8f acc = {};
#pragma unroll
    for (int kk = 0; kk < 4; ++kk) {
      const v16bf bf = *(const v16bf*)(W1 + (((kk * 8 + nn) * 32 + lane) << 4));
      acc = __builtin_amdgcn_wmma_f32_16x16x32_bf16(false, a[kk], false, bf,
                                                    (short)0, acc, false, false);
    }
    const int   n    = nn * 16 + row;
    const float bias = B1[n];
#pragma unroll
    for (int v = 0; v < 8; ++v)
      act[wave][v + half][n] = (__bf16)gelu_fast_exact(acc[v] + bias);
  }
  __builtin_amdgcn_wave_barrier();                  // wave-private LDS: order only

  // reload layer-1 activations as layer-2 A fragments (ds_load_b128 pairs)
#pragma unroll
  for (int kk = 0; kk < 4; ++kk) {
    const int kb = kk * 32 + half;
    V16Q u;
    u.q[0] = *(const uint4*)&act[wave][row][kb];
    u.q[1] = *(const uint4*)&act[wave][row][kb + 16];
    a[kk]  = u.v;
  }

  // per-row scatter metadata for the 8 M-rows this lane owns in C
  float ew[8]; int tgt[8];
#pragma unroll
  for (int v = 0; v < 8; ++v) {
    int e = eb + v + half;
    if (e >= nEdges) e = nEdges - 1;
    ew[v]  = edge_weights[e];
    tgt[v] = edges[e];                              // edges[0][e]
  }

  // layer 2 + edge weighting + atomic scatter into segment sums
#pragma unroll
  for (int nn = 0; nn < 8; ++nn) {
    v8f acc = {};
#pragma unroll
    for (int kk = 0; kk < 4; ++kk) {
      const v16bf bf = *(const v16bf*)(W2 + (((kk * 8 + nn) * 32 + lane) << 4));
      acc = __builtin_amdgcn_wmma_f32_16x16x32_bf16(false, a[kk], false, bf,
                                                    (short)0, acc, false, false);
    }
    const int   n    = nn * 16 + row;
    const float bias = B2[n];
#pragma unroll
    for (int v = 0; v < 8; ++v) {
      const float msg = gelu_fast_exact(acc[v] + bias) * ew[v];
      unsafeAtomicAdd(&sums[(size_t)tgt[v] * DIM + n], msg);
    }
  }
  if (lane < 16 && (eb + row) < nEdges)
    unsafeAtomicAdd(&counts[edges[eb + row]], 1.0f);
}

// ---------------------------------------------------------------------------
// Update kernel: one wave per 16-node tile, K=256 (node_reps ++ mean-agg).
// ---------------------------------------------------------------------------
__global__ __launch_bounds__(128) void upd_kernel(
    const float* __restrict__ node_reps, const float* __restrict__ sums,
    const float* __restrict__ counts,
    const __bf16* __restrict__ W1, const float* __restrict__ B1,
    const __bf16* __restrict__ W2, const float* __restrict__ B2,
    float* __restrict__ out, int nNodes) {
  __shared__ __align__(16) __bf16 act[4][16][DIM];
  const int wave = threadIdx.x >> 5;
  const int lane = threadIdx.x & 31;
  const int nb   = (blockIdx.x * 4 + wave) * 16;
  if (nb >= nNodes) return;
  const int half = (lane < 16) ? 0 : 8;
  const int row  = lane & 15;
  const int node = (nb + row < nNodes) ? (nb + row) : (nNodes - 1);

  __builtin_prefetch(W1 + (lane << 6), 0, 1);
  __builtin_prefetch(W2 + (lane << 6), 0, 1);

  const float inv = __builtin_amdgcn_rcpf(fmaxf(counts[node], 1.0f));
  const float* __restrict__ srcN = node_reps + (size_t)node * DIM;
  const float* __restrict__ srcS = sums + (size_t)node * DIM;

  v16bf a[8];
#pragma unroll
  for (int kk = 0; kk < 4; ++kk) {                  // h[:,0:128] = node_reps
    const int kb = kk * 32 + half;
#pragma unroll
    for (int i = 0; i < 8; ++i) {
      a[kk][i]     = (__bf16)srcN[kb + i];
      a[kk][8 + i] = (__bf16)srcN[kb + 16 + i];
    }
  }
#pragma unroll
  for (int kk = 0; kk < 4; ++kk) {                  // h[:,128:256] = mean agg
    const int kb = kk * 32 + half;
#pragma unroll
    for (int i = 0; i < 8; ++i) {
      a[4 + kk][i]     = (__bf16)(srcS[kb + i]      * inv);
      a[4 + kk][8 + i] = (__bf16)(srcS[kb + 16 + i] * inv);
    }
  }

  // layer 1: [16x256] @ W1'[256x128]
#pragma unroll
  for (int nn = 0; nn < 8; ++nn) {
    v8f acc = {};
#pragma unroll
    for (int kk = 0; kk < 8; ++kk) {
      const v16bf bf = *(const v16bf*)(W1 + (((kk * 8 + nn) * 32 + lane) << 4));
      acc = __builtin_amdgcn_wmma_f32_16x16x32_bf16(false, a[kk], false, bf,
                                                    (short)0, acc, false, false);
    }
    const int   n    = nn * 16 + row;
    const float bias = B1[n];
#pragma unroll
    for (int v = 0; v < 8; ++v)
      act[wave][v + half][n] = (__bf16)gelu_fast_exact(acc[v] + bias);
  }
  __builtin_amdgcn_wave_barrier();

#pragma unroll
  for (int kk = 0; kk < 4; ++kk) {
    const int kb = kk * 32 + half;
    V16Q u;
    u.q[0] = *(const uint4*)&act[wave][row][kb];
    u.q[1] = *(const uint4*)&act[wave][row][kb + 16];
    a[kk]  = u.v;
  }

  // layer 2 + final f32 store
#pragma unroll
  for (int nn = 0; nn < 8; ++nn) {
    v8f acc = {};
#pragma unroll
    for (int kk = 0; kk < 4; ++kk) {
      const v16bf bf = *(const v16bf*)(W2 + (((kk * 8 + nn) * 32 + lane) << 4));
      acc = __builtin_amdgcn_wmma_f32_16x16x32_bf16(false, a[kk], false, bf,
                                                    (short)0, acc, false, false);
    }
    const int   n    = nn * 16 + row;
    const float bias = B2[n];
#pragma unroll
    for (int v = 0; v < 8; ++v) {
      const int m = nb + v + half;
      if (m < nNodes) out[(size_t)m * DIM + n] = gelu_fast_exact(acc[v] + bias);
    }
  }
}

// ---------------------------------------------------------------------------
// Host launcher. d_in order (insertion order of setup_inputs, params flattened
// prepare[L0,L1], update[L0,L1], each (gamma,beta,mu,var,w,b)):
//   0 node_reps | 1 edges | 2 edge_weights | 3.. params (24 arrays)
// ---------------------------------------------------------------------------
extern "C" void kernel_launch(void* const* d_in, const int* in_sizes, int n_in,
                              void* d_out, int out_size, void* d_ws, size_t ws_size,
                              hipStream_t stream) {
  const float* node_reps = (const float*)d_in[0];
  const int*   edges     = (const int*)d_in[1];
  const float* eweights  = (const float*)d_in[2];
  const int N = in_sizes[0] / DIM;
  const int E = in_sizes[2];

  char* ws = (char*)d_ws;
  __bf16* Wp[4] = { (__bf16*)(ws + 0),      (__bf16*)(ws + 32768),
                    (__bf16*)(ws + 65536),  (__bf16*)(ws + 131072) };
  float*  Bp[4] = { (float*)(ws + 163840),  (float*)(ws + 164352),
                    (float*)(ws + 164864),  (float*)(ws + 165376) };
  float* counts = (float*)(ws + 165888);                 // N floats
  float* sums   = (float*)(ws + 165888 + ((size_t)N * 4 + 255) / 256 * 256);

  static const int pbase[4] = {3, 9, 15, 21};
  static const int din[4]   = {128, 128, 256, 128};
  for (int l = 0; l < 4; ++l) {
    pack_kernel<<<1, 128, 0, stream>>>(
        (const float*)d_in[pbase[l] + 0], (const float*)d_in[pbase[l] + 1],
        (const float*)d_in[pbase[l] + 2], (const float*)d_in[pbase[l] + 3],
        (const float*)d_in[pbase[l] + 4], (const float*)d_in[pbase[l] + 5],
        din[l], Wp[l], Bp[l]);
  }
  zero_kernel<<<2048, 256, 0, stream>>>(sums, counts, (size_t)N * DIM, N);

  const int etiles = (E + 15) / 16;
  msg_kernel<<<(etiles + 3) / 4, 128, 0, stream>>>(
      node_reps, edges, eweights, Wp[0], Bp[0], Wp[1], Bp[1], sums, counts, E);

  const int ntiles = (N + 15) / 16;
  upd_kernel<<<(ntiles + 3) / 4, 128, 0, stream>>>(
      node_reps, sums, counts, Wp[2], Bp[2], Wp[3], Bp[3], (float*)d_out, N);
}
